// SparseNet_17489106830016
// MI455X (gfx1250) — compile-verified, matrix operand order checked by reading, not run on hardware
//
#include <hip/hip_runtime.h>

// ---------------------------------------------------------------------------
// CDNA5 (gfx1250) sparse 3D conv stack via V_WMMA_F32_16X16X32_BF16.
//   layer: 3x3x3 subm-conv (gather-GEMM over 27 taps) + bias + BN + ReLU
//   channels: 16 -> 64 -> 128 -> 64, then scatter to dense NCDHW output.
// Wave32; one wave owns a 16-row tile and all Cout tiles.
// ---------------------------------------------------------------------------

typedef __bf16  vbf16 __attribute__((ext_vector_type(16)));
typedef __bf16  vbf8  __attribute__((ext_vector_type(8)));
typedef float   vf8   __attribute__((ext_vector_type(8)));

#define D3 1728   // 12*12*12
#define BATCH 512

// ---------------- grid map ----------------
__global__ void grid_init_kernel(int* __restrict__ grid, int total) {
    int i = blockIdx.x * blockDim.x + threadIdx.x;
    if (i < total) grid[i] = -1;
}

__global__ void grid_scatter_kernel(int* __restrict__ grid,
                                    const int* __restrict__ idx, int N) {
    int n = blockIdx.x * blockDim.x + threadIdx.x;
    if (n >= N) return;
    int4 ii = ((const int4*)idx)[n];            // b,z,y,x
    grid[((ii.x * 12 + ii.y) * 12 + ii.z) * 12 + ii.w] = n;
}

// ---------------- input f32 -> bf16 (pad 16 -> 32 channels) ----------------
__global__ void convert_in_kernel(const float* __restrict__ feats,
                                  __bf16* __restrict__ f0, int N) {
    int i = blockIdx.x * blockDim.x + threadIdx.x;
    if (i >= N * 32) return;
    int n = i >> 5, c = i & 31;
    f0[i] = (c < 16) ? (__bf16)feats[n * 16 + c] : (__bf16)0.0f;
}

// ---------------- weight prep: (3,3,3,CINR,COUT) f32 -> fragment-order bf16
// wp[tap][kc][ct][lane][e]; element e of lane maps to k = kc*32 + hh*8 + (e<8?e:e+8)
template<int CINR, int CIN, int COUT>
__global__ void wprep_kernel(const float* __restrict__ W, __bf16* __restrict__ wp) {
    constexpr int KC = CIN / 32, CT = COUT / 16;
    int i = blockIdx.x * blockDim.x + threadIdx.x;
    if (i >= 27 * KC * CT * 512) return;
    int e    = i & 15;
    int lane = (i >> 4) & 31;
    int rest = i >> 9;
    int ct  = rest % CT;  rest /= CT;
    int kc  = rest % KC;
    int tap = rest / KC;
    int hh  = (lane >> 4) & 1;
    int col = lane & 15;
    int k    = kc * 32 + hh * 8 + (e < 8 ? e : e + 8);
    int cout = ct * 16 + col;
    float v = (k < CINR) ? W[((long)tap * CINR + k) * COUT + cout] : 0.0f;
    wp[i] = (__bf16)v;
}

// ---------------- gather-GEMM conv (+bias, +BN stats) ----------------
// block = 128 threads = 4 waves; each wave does 4 row-tiles of 16 rows.
// fin : [N+1][CIN] bf16 (row N is all-zero)     fout: [N][COUT] bf16 pre-act
template<int CIN, int COUT>
__launch_bounds__(128)
__global__ void conv_kernel(const __bf16* __restrict__ fin,
                            const __bf16* __restrict__ wp,
                            const float*  __restrict__ bias,
                            const int*    __restrict__ idx,
                            const int*    __restrict__ grid,
                            __bf16*       __restrict__ fout,
                            float* __restrict__ gsum,
                            float* __restrict__ gsumsq,
                            int N) {
    constexpr int KC = CIN / 32, CT = COUT / 16;
    __shared__ float lsum[COUT];
    __shared__ float lsq [COUT];
    int t = threadIdx.x;
    for (int c = t; c < COUT; c += 128) { lsum[c] = 0.f; lsq[c] = 0.f; }
    __syncthreads();

    int wave = t >> 5;
    int lane = t & 31;
    int hh   = (lane >> 4) & 1;     // half-wave id
    int col  = lane & 15;

    for (int rt = 0; rt < 4; ++rt) {
        int rowbase = (blockIdx.x * 16 + wave * 4 + rt) * 16;   // 16-row tile
        int row = rowbase + col;
        int4 ii = ((const int4*)idx)[row];                       // b,z,y,x

        vf8 acc[CT];
        #pragma unroll
        for (int ct = 0; ct < CT; ++ct)
            #pragma unroll
            for (int e = 0; e < 8; ++e) acc[ct][e] = 0.0f;

        for (int tap = 0; tap < 27; ++tap) {
            int dz = tap / 9 - 1, dy = (tap / 3) % 3 - 1, dx = tap % 3 - 1;
            int zz = ii.y + dz, yy = ii.z + dy, xx = ii.w + dx;
            bool ok = ((unsigned)zz < 12u) & ((unsigned)yy < 12u) & ((unsigned)xx < 12u);
            int gi = ok ? (((ii.x * 12 + zz) * 12 + yy) * 12 + xx) : 0;
            int g  = grid[gi];
            int r  = (ok && g >= 0) ? g : N;                    // row N == zeros
            const __bf16* fr = fin + (long)r * CIN;

            #pragma unroll
            for (int kc = 0; kc < KC; ++kc) {
                // A-fragment: two contiguous 16B chunks per lane (documented layout)
                vbf8 lo = *(const vbf8*)(fr + kc * 32 + hh * 8);
                vbf8 hi = *(const vbf8*)(fr + kc * 32 + 16 + hh * 8);
                vbf16 a = __builtin_shufflevector(lo, hi,
                            0,1,2,3,4,5,6,7,8,9,10,11,12,13,14,15);
                const __bf16* wb = wp + (long)((tap * KC + kc) * CT) * 512 + lane * 16;
                #pragma unroll
                for (int ct = 0; ct < CT; ++ct) {
                    vbf16 bfrag = *(const vbf16*)(wb + ct * 512);
                    acc[ct] = __builtin_amdgcn_wmma_f32_16x16x32_bf16(
                        false, a, false, bfrag, (short)0, acc[ct], false, false);
                }
            }
        }

        // bias, store bf16 pre-activation, accumulate BN stats
        #pragma unroll
        for (int ct = 0; ct < CT; ++ct) {
            int ch = ct * 16 + col;
            float bz = bias[ch];
            float s = 0.f, s2 = 0.f;
            #pragma unroll
            for (int e = 0; e < 8; ++e) {
                float v = acc[ct][e] + bz;
                int orow = rowbase + hh * 8 + e;                // C/D layout: M = e + hh*8
                fout[(long)orow * COUT + ch] = (__bf16)v;
                s += v; s2 += v * v;
            }
            atomicAdd(&lsum[ch], s);
            atomicAdd(&lsq [ch], s2);
        }
    }
    __syncthreads();
    for (int c = t; c < COUT; c += 128) {
        atomicAdd(&gsum[c],   lsum[c]);
        atomicAdd(&gsumsq[c], lsq[c]);
    }
}

// ---------------- fold BN stats to scale/shift ----------------
__global__ void stats_kernel(const float* __restrict__ gsum,
                             const float* __restrict__ gsumsq,
                             const float* __restrict__ gamma,
                             const float* __restrict__ beta,
                             float* __restrict__ scale,
                             float* __restrict__ shift,
                             int Cout, float invN) {
    int c = threadIdx.x;
    if (c >= Cout) return;
    float mu  = gsum[c] * invN;
    float var = gsumsq[c] * invN - mu * mu;
    float sc  = gamma[c] * rsqrtf(var + 1e-5f);
    scale[c] = sc;
    shift[c] = beta[c] - mu * sc;
}

// ---------------- BN+ReLU in place (bf16), feeds next layer ----------------
__global__ void apply_kernel(__bf16* __restrict__ f,
                             const float* __restrict__ scale,
                             const float* __restrict__ shift,
                             int total, int cmask) {
    int i = blockIdx.x * blockDim.x + threadIdx.x;
    if (i >= total) return;
    int c = i & cmask;
    float v = (float)f[i] * scale[c] + shift[c];
    f[i] = (__bf16)fmaxf(v, 0.0f);
}

// ---------------- final BN+ReLU + scatter to dense NCDHW f32 ----------------
__global__ void scatter_out_kernel(const __bf16* __restrict__ f2,
                                   const float* __restrict__ scale,
                                   const float* __restrict__ shift,
                                   const int* __restrict__ idx,
                                   float* __restrict__ out, int N) {
    int i = blockIdx.x * blockDim.x + threadIdx.x;
    if (i >= N * 64) return;
    int n = i >> 6, c = i & 63;
    float v = (float)f2[i] * scale[c] + shift[c];
    v = fmaxf(v, 0.0f);
    int4 ii = ((const int4*)idx)[n];
    out[((long)ii.x * 64 + c) * D3 + ii.y * 144 + ii.z * 12 + ii.w] = v;
}

// ---------------------------------------------------------------------------
extern "C" void kernel_launch(void* const* d_in, const int* in_sizes, int n_in,
                              void* d_out, int out_size, void* d_ws, size_t ws_size,
                              hipStream_t stream) {
    const float* feats  = (const float*)d_in[0];
    const int*   idx    = (const int*)  d_in[1];
    const float* W0 = (const float*)d_in[3],  *b0 = (const float*)d_in[4];
    const float* g0 = (const float*)d_in[5],  *be0 = (const float*)d_in[6];
    const float* W1 = (const float*)d_in[7],  *b1 = (const float*)d_in[8];
    const float* g1 = (const float*)d_in[9],  *be1 = (const float*)d_in[10];
    const float* W2 = (const float*)d_in[11], *b2 = (const float*)d_in[12];
    const float* g2 = (const float*)d_in[13], *be2 = (const float*)d_in[14];
    float* out = (float*)d_out;

    const int N = in_sizes[1] / 4;            // 442368 (multiple of 256)
    const int B = BATCH;

    // ---- workspace carve-up (256B aligned) ----
    char* ws = (char*)d_ws;
    size_t off = 0;
    auto carve = [&](size_t bytes) -> char* {
        char* p = ws + off;
        off = (off + bytes + 255) & ~(size_t)255;
        return p;
    };
    int*    grid  = (int*)   carve((size_t)B * D3 * 4);
    __bf16* fA    = (__bf16*)carve(((size_t)N * 128 + 128) * 2);   // ping
    __bf16* fB    = (__bf16*)carve(((size_t)N * 128 + 128) * 2);   // pong
    __bf16* wp0   = (__bf16*)carve((size_t)27 * 1 * 4 * 512 * 2);
    __bf16* wp1   = (__bf16*)carve((size_t)27 * 2 * 8 * 512 * 2);
    __bf16* wp2   = (__bf16*)carve((size_t)27 * 4 * 4 * 512 * 2);
    float*  stats = (float*) carve(4 * 128 * 4);
    float* gsum = stats, *gsq = stats + 128, *scale = stats + 256, *shift = stats + 384;
    (void)ws_size; (void)n_in;

    const int nwg = N / 256;                  // 1728 conv workgroups

    // ---- setup: grid map, input convert, weight permute ----
    grid_init_kernel<<<(B * D3 + 255) / 256, 256, 0, stream>>>(grid, B * D3);
    grid_scatter_kernel<<<(N + 255) / 256, 256, 0, stream>>>(grid, idx, N);
    convert_in_kernel<<<(N * 32 + 255) / 256, 256, 0, stream>>>(feats, fA, N);
    wprep_kernel<16, 32, 64>  <<<(27 * 1 * 4 * 512 + 255) / 256, 256, 0, stream>>>(W0, wp0);
    wprep_kernel<64, 64, 128> <<<(27 * 2 * 8 * 512 + 255) / 256, 256, 0, stream>>>(W1, wp1);
    wprep_kernel<128, 128, 64><<<(27 * 4 * 4 * 512 + 255) / 256, 256, 0, stream>>>(W2, wp2);

    // ---- layer 0: 32(pad of 16) -> 64, fA -> fB ----
    hipMemsetAsync(stats, 0, 2 * 128 * 4, stream);
    hipMemsetAsync(fA + (size_t)N * 32, 0, 32 * 2, stream);         // zero row
    conv_kernel<32, 64><<<nwg, 128, 0, stream>>>(fA, wp0, b0, idx, grid, fB, gsum, gsq, N);
    stats_kernel<<<1, 64, 0, stream>>>(gsum, gsq, g0, be0, scale, shift, 64, 1.0f / N);
    apply_kernel<<<((size_t)N * 64 + 255) / 256, 256, 0, stream>>>(fB, scale, shift, N * 64, 63);

    // ---- layer 1: 64 -> 128, fB -> fA ----
    hipMemsetAsync(stats, 0, 2 * 128 * 4, stream);
    hipMemsetAsync(fB + (size_t)N * 64, 0, 64 * 2, stream);         // zero row
    conv_kernel<64, 128><<<nwg, 128, 0, stream>>>(fB, wp1, b1, idx, grid, fA, gsum, gsq, N);
    stats_kernel<<<1, 128, 0, stream>>>(gsum, gsq, g1, be1, scale, shift, 128, 1.0f / N);
    apply_kernel<<<((size_t)N * 128 + 255) / 256, 256, 0, stream>>>(fA, scale, shift, N * 128, 127);

    // ---- layer 2: 128 -> 64, fA -> fB ----
    hipMemsetAsync(stats, 0, 2 * 128 * 4, stream);
    hipMemsetAsync(fA + (size_t)N * 128, 0, 128 * 2, stream);       // zero row
    conv_kernel<128, 64><<<nwg, 128, 0, stream>>>(fA, wp2, b2, idx, grid, fB, gsum, gsq, N);
    stats_kernel<<<1, 64, 0, stream>>>(gsum, gsq, g2, be2, scale, shift, 64, 1.0f / N);

    // ---- final: zero dense output, BN+ReLU+scatter ----
    hipMemsetAsync(out, 0, (size_t)out_size * 4, stream);
    scatter_out_kernel<<<((size_t)N * 64 + 255) / 256, 256, 0, stream>>>(fB, scale, shift, idx, out, N);
}
